// TransformerV1_38216618999839
// MI455X (gfx1250) — matrix operand order repaired and checked by reading, not compiled, not run on hardware
//
#include <hip/hip_runtime.h>
#include <hip/hip_bf16.h>
#include <cmath>

// ---------------------------------------------------------------------------
// Types for CDNA5 WMMA
// ---------------------------------------------------------------------------
typedef __attribute__((ext_vector_type(8)))  __bf16 v8bf;
typedef __attribute__((ext_vector_type(16))) __bf16 v16bf;
typedef __attribute__((ext_vector_type(8)))  float  v8f;

#define TM 64
#define TN 64
#define TKDIM 32
#define LDT 40   // padded LDS row stride in bf16 elems (40*2=80B, 16B aligned rows)

// Problem constants (match reference)
#define CV 16384
#define CE 768
#define CH 3072
#define CB 4
#define CS 512
#define CL 5
#define MR (CB * CS)   // 2048 rows of activations

// ---------------------------------------------------------------------------
// Small device helpers
// ---------------------------------------------------------------------------
__device__ __forceinline__ void cvt_store8(__bf16* dst, float4 f0, float4 f1)
{
    dst[0] = (__bf16)f0.x; dst[1] = (__bf16)f0.y;
    dst[2] = (__bf16)f0.z; dst[3] = (__bf16)f0.w;
    dst[4] = (__bf16)f1.x; dst[5] = (__bf16)f1.y;
    dst[6] = (__bf16)f1.z; dst[7] = (__bf16)f1.w;
}

__device__ __forceinline__ void cvt_store8_strided(__bf16* base, int stride, float4 f0, float4 f1)
{
    const float v[8] = {f0.x, f0.y, f0.z, f0.w, f1.x, f1.y, f1.z, f1.w};
#pragma unroll
    for (int i = 0; i < 8; ++i) base[i * stride] = (__bf16)v[i];
}

// A fragment (16x32 bf16): lane<16 holds K 0..15, lane>=16 holds K 16..31 (contiguous)
__device__ __forceinline__ v16bf frag_a(const __bf16* p)
{
    const v8bf lo = *(const v8bf*)(p);
    const v8bf hi = *(const v8bf*)(p + 8);
    return __builtin_shufflevector(lo, hi,
        0, 1, 2, 3, 4, 5, 6, 7, 8, 9, 10, 11, 12, 13, 14, 15);
}

// B fragment (32x16 bf16): lane<16 holds K 0..15, lane>=16 holds K 16..31 (contiguous)
__device__ __forceinline__ v16bf frag_b(const __bf16* p)
{
    const v8bf lo = *(const v8bf*)(p);
    const v8bf hi = *(const v8bf*)(p + 8);
    return __builtin_shufflevector(lo, hi,
        0, 1, 2, 3, 4, 5, 6, 7, 8, 9, 10, 11, 12, 13, 14, 15);
}

// ---------------------------------------------------------------------------
// Generic WMMA GEMM:  C = act(alpha * (A @ B) + bias)
//   A: M x K (row major)
//   B_IS_KN=true : B is K x N (weights)          -> transposed into LDS [n][k]
//   B_IS_KN=false: B is N x K (compute A @ B^T)  -> copied into LDS [n][k]
// Batched over gridDim.z with element strides.
// 256 threads = 8 waves; block tile 64x64; wave tile 16x32 (two 16x16 accs).
// Double-buffered LDS: one barrier per K step; next tile's global loads are
// issued before the current tile's WMMAs so HBM fetch overlaps matrix math.
// ---------------------------------------------------------------------------
template <bool B_IS_KN, bool RELU>
__global__ __launch_bounds__(256) void gemm_wmma_kernel(
    const float* __restrict__ A, const float* __restrict__ Bm,
    const float* __restrict__ bias, float* __restrict__ C,
    int M, int N, int K,
    long long strideA, long long strideB, long long strideC,
    float alpha)
{
    __shared__ __align__(16) __bf16 sA[2][TM * LDT];
    __shared__ __align__(16) __bf16 sB[2][TN * LDT];

    const int z = blockIdx.z;
    A  += (long long)z * strideA;
    Bm += (long long)z * strideB;
    C  += (long long)z * strideC;

    const int bn = blockIdx.x * TN;
    const int bm = blockIdx.y * TM;

    const int tid  = threadIdx.x;
    const int lane = tid & 31;
    const int wave = tid >> 5;     // 0..7
    const int wm   = wave & 3;     // 0..3 -> +16*wm in M
    const int wn   = wave >> 2;    // 0..1 -> +32*wn in N
    const int l16  = lane & 15;
    const int hlf  = lane >> 4;    // 0 or 1

    // ---- per-thread staging coordinates ----
    // A: each thread owns 8 consecutive K of one row
    const int  ar   = tid >> 2;            // 0..63
    const int  akq  = (tid & 3) * 8;       // 0,8,16,24
    const int  agm  = bm + ar;
    const bool aok  = agm < M;
    const float* Ap = A + (long long)agm * K + akq;
    // B: coords depend on layout
    const int  bkk  = tid >> 3;            // KN: 0..31
    const int  bnq  = (tid & 7) * 8;       // KN: 0..56
    const int  bnr  = tid >> 2;            // NK: n local 0..63
    const int  bkq  = (tid & 3) * 8;       // NK: 0,8,16,24
    const float* Bp = B_IS_KN
        ? (Bm + (long long)bkk * N + (bn + bnq))     // advance by k0*N per step
        : (Bm + (long long)(bn + bnr) * K + bkq);    // advance by k0 per step

    v8f acc0 = {};
    v8f acc1 = {};

    // ---- prologue: stage tile k0 = 0 into buffer 0 ----
    float4 a0 = {0.f,0.f,0.f,0.f}, a1 = {0.f,0.f,0.f,0.f};
    if (aok) {
        const float4* p = (const float4*)(Ap);
        a0 = p[0]; a1 = p[1];
    }
    float4 b0, b1;
    {
        const float4* p = (const float4*)(Bp);
        b0 = p[0]; b1 = p[1];
    }
    cvt_store8(&sA[0][ar * LDT + akq], a0, a1);
    if (B_IS_KN) cvt_store8_strided(&sB[0][bnq * LDT + bkk], LDT, b0, b1);
    else         cvt_store8(&sB[0][bnr * LDT + bkq], b0, b1);

    int cur = 0;
    for (int k0 = 0; k0 < K; k0 += TKDIM) {
        __syncthreads();   // staged tile `cur` visible; prior reads of cur^1 done

        const bool more = (k0 + TKDIM) < K;
        // ---- issue next tile's global loads (overlaps with WMMAs below) ----
        float4 na0 = {0.f,0.f,0.f,0.f}, na1 = {0.f,0.f,0.f,0.f};
        float4 nb0, nb1;
        if (more) {
            const int kn = k0 + TKDIM;
            if (aok) {
                const float4* p = (const float4*)(Ap + kn);
                na0 = p[0]; na1 = p[1];
            }
            const float4* p = B_IS_KN
                ? (const float4*)(Bp + (long long)kn * N)
                : (const float4*)(Bp + kn);
            nb0 = p[0]; nb1 = p[1];
        }

        // ---- compute on buffer `cur` ----
        const __bf16* sAc = sA[cur];
        const __bf16* sBc = sB[cur];
        const v16bf af  = frag_a(&sAc[(wm * 16 + l16) * LDT + hlf * 16]);
        const v16bf bf0 = frag_b(&sBc[(wn * 32 + l16) * LDT + hlf * 16]);
        const v16bf bf1 = frag_b(&sBc[(wn * 32 + 16 + l16) * LDT + hlf * 16]);

        acc0 = __builtin_amdgcn_wmma_f32_16x16x32_bf16(
            false, af, false, bf0, (short)0, acc0, false, false);
        acc1 = __builtin_amdgcn_wmma_f32_16x16x32_bf16(
            false, af, false, bf1, (short)0, acc1, false, false);

        // ---- stage next tile into the other buffer ----
        if (more) {
            const int nxt = cur ^ 1;
            cvt_store8(&sA[nxt][ar * LDT + akq], na0, na1);
            if (B_IS_KN) cvt_store8_strided(&sB[nxt][bnq * LDT + bkk], LDT, nb0, nb1);
            else         cvt_store8(&sB[nxt][bnr * LDT + bkq], nb0, nb1);
            cur = nxt;
        }
    }

    // ---- epilogue: C/D layout — lane n = l16, VGPR r -> m = r + 8*hlf ----
#pragma unroll
    for (int sub = 0; sub < 2; ++sub) {
        const v8f acc = sub ? acc1 : acc0;
        const int gn = bn + wn * 32 + sub * 16 + l16;
        const float bv = bias ? bias[gn] : 0.0f;
#pragma unroll
        for (int r = 0; r < 8; ++r) {
            const int gm = bm + wm * 16 + r + 8 * hlf;
            if (gm < M) {
                float v = alpha * acc[r] + bv;
                if (RELU) v = v > 0.0f ? v : 0.0f;
                C[(long long)gm * N + gn] = v;
            }
        }
    }
}

// ---------------------------------------------------------------------------
// y = LayerNorm(x + r) * g + b      one block (256 thr) per row, E <= 1024
// ---------------------------------------------------------------------------
__global__ __launch_bounds__(256) void ln_residual_kernel(
    const float* __restrict__ x, const float* __restrict__ r,
    const float* __restrict__ g, const float* __restrict__ b,
    float* __restrict__ y, int E_)
{
    __shared__ float red[256];
    const long long row = blockIdx.x;
    const float* xr = x + row * E_;
    const float* rr = r + row * E_;
    float* yr = y + row * E_;
    const int tid = threadIdx.x;

    float v[4];
    int ne = 0;
    float s = 0.0f;
    for (int e = tid; e < E_; e += 256) {
        const float t = xr[e] + rr[e];
        v[ne++] = t;
        s += t;
    }
    red[tid] = s;
    __syncthreads();
    for (int off = 128; off > 0; off >>= 1) {
        if (tid < off) red[tid] += red[tid + off];
        __syncthreads();
    }
    const float mean = red[0] / (float)E_;
    __syncthreads();

    s = 0.0f;
    for (int i = 0; i < ne; ++i) {
        const float d = v[i] - mean;
        s += d * d;
    }
    red[tid] = s;
    __syncthreads();
    for (int off = 128; off > 0; off >>= 1) {
        if (tid < off) red[tid] += red[tid + off];
        __syncthreads();
    }
    const float rstd = rsqrtf(red[0] / (float)E_ + 1e-5f);

    int i = 0;
    for (int e = tid; e < E_; e += 256) {
        yr[e] = (v[i] - mean) * rstd * g[e] + b[e];
        ++i;
    }
}

// ---------------------------------------------------------------------------
// In-place row softmax: one block per row, arbitrary width
// ---------------------------------------------------------------------------
__global__ __launch_bounds__(256) void softmax_rows_kernel(float* __restrict__ x, int W)
{
    __shared__ float red[256];
    const long long row = blockIdx.x;
    float* xr = x + row * (long long)W;
    const int tid = threadIdx.x;

    float m = -3.4e38f;
    for (int e = tid; e < W; e += 256) m = fmaxf(m, xr[e]);
    red[tid] = m;
    __syncthreads();
    for (int off = 128; off > 0; off >>= 1) {
        if (tid < off) red[tid] = fmaxf(red[tid], red[tid + off]);
        __syncthreads();
    }
    m = red[0];
    __syncthreads();

    float s = 0.0f;
    for (int e = tid; e < W; e += 256) {
        const float t = __expf(xr[e] - m);
        xr[e] = t;
        s += t;
    }
    red[tid] = s;
    __syncthreads();
    for (int off = 128; off > 0; off >>= 1) {
        if (tid < off) red[tid] += red[tid + off];
        __syncthreads();
    }
    const float inv = 1.0f / red[0];
    for (int e = tid; e < W; e += 256) xr[e] *= inv;
}

// Gather last token per batch: hN[b*E+e] = x[(b*S + S-1)*E + e]
__global__ __launch_bounds__(256) void gather_last_kernel(
    const float* __restrict__ x, float* __restrict__ hN, int S_, int E_)
{
    const int idx = blockIdx.x * 256 + threadIdx.x;
    if (idx >= CB * E_) return;
    const int b = idx / E_;
    const int e = idx - b * E_;
    hN[idx] = x[((long long)b * S_ + (S_ - 1)) * E_ + e];
}

// ---------------------------------------------------------------------------
// Host-side launch helper
// ---------------------------------------------------------------------------
static inline void launch_gemm(hipStream_t st,
    const float* A, const float* Bm, const float* bias, float* C,
    int M, int N, int K, int batch,
    long long sA, long long sB, long long sC,
    float alpha, bool relu, bool b_kn)
{
    dim3 grid(N / TN, (M + TM - 1) / TM, batch);
    dim3 block(256);
    if (b_kn) {
        if (relu) gemm_wmma_kernel<true,  true ><<<grid, block, 0, st>>>(A, Bm, bias, C, M, N, K, sA, sB, sC, alpha);
        else      gemm_wmma_kernel<true,  false><<<grid, block, 0, st>>>(A, Bm, bias, C, M, N, K, sA, sB, sC, alpha);
    } else {
        if (relu) gemm_wmma_kernel<false, true ><<<grid, block, 0, st>>>(A, Bm, bias, C, M, N, K, sA, sB, sC, alpha);
        else      gemm_wmma_kernel<false, false><<<grid, block, 0, st>>>(A, Bm, bias, C, M, N, K, sA, sB, sC, alpha);
    }
}

extern "C" void kernel_launch(void* const* d_in, const int* in_sizes, int n_in,
                              void* d_out, int out_size, void* d_ws, size_t ws_size,
                              hipStream_t stream)
{
    (void)in_sizes; (void)n_in; (void)out_size; (void)ws_size;

    // ---- inputs (setup_inputs dict order, nested dicts flattened) ----
    const float* seq       = (const float*)d_in[0];   // (B*S, V)
    const float* embed_w   = (const float*)d_in[1];   // (V, E)
    const float* embed_b   = (const float*)d_in[2];   // (E)
    // enc: wk bk wq bq wv bv w1 b1 w2 b2 ln1_g ln1_b ln2_g ln2_b
    const float* e_wk   = (const float*)d_in[3];
    const float* e_bk   = (const float*)d_in[4];
    const float* e_wq   = (const float*)d_in[5];
    const float* e_bq   = (const float*)d_in[6];
    const float* e_wv   = (const float*)d_in[7];
    const float* e_bv   = (const float*)d_in[8];
    const float* e_w1   = (const float*)d_in[9];
    const float* e_b1   = (const float*)d_in[10];
    const float* e_w2   = (const float*)d_in[11];
    const float* e_b2   = (const float*)d_in[12];
    const float* e_ln1g = (const float*)d_in[13];
    const float* e_ln1b = (const float*)d_in[14];
    const float* e_ln2g = (const float*)d_in[15];
    const float* e_ln2b = (const float*)d_in[16];
    // dec: wk bk wq bq wv bv wk2 bk2 wq2 bq2 wv2 bv2 w1 b1 w2 b2 ln1_g ln1_b ln2_g ln2_b ln3_g ln3_b
    const float* d_wk   = (const float*)d_in[17];
    const float* d_bk   = (const float*)d_in[18];
    const float* d_wq   = (const float*)d_in[19];
    const float* d_bq   = (const float*)d_in[20];
    const float* d_wv   = (const float*)d_in[21];
    const float* d_bv   = (const float*)d_in[22];
    const float* d_wk2  = (const float*)d_in[23];
    const float* d_bk2  = (const float*)d_in[24];
    const float* d_wq2  = (const float*)d_in[25];
    const float* d_bq2  = (const float*)d_in[26];
    const float* d_wv2  = (const float*)d_in[27];
    const float* d_bv2  = (const float*)d_in[28];
    const float* d_w1   = (const float*)d_in[29];
    const float* d_b1   = (const float*)d_in[30];
    const float* d_w2   = (const float*)d_in[31];
    const float* d_b2   = (const float*)d_in[32];
    const float* d_ln1g = (const float*)d_in[33];
    const float* d_ln1b = (const float*)d_in[34];
    const float* d_ln2g = (const float*)d_in[35];
    const float* d_ln2b = (const float*)d_in[36];
    const float* d_ln3g = (const float*)d_in[37];
    const float* d_ln3b = (const float*)d_in[38];
    const float* unembed_w = (const float*)d_in[39];  // (E, V)
    const float* unembed_b = (const float*)d_in[40];  // (V)

    // ---- workspace layout (floats) ----
    float* ws = (float*)d_ws;
    const long long ME = (long long)MR * CE;    // 2048*768
    float* embeds = ws + 0 * ME;
    float* t0     = ws + 1 * ME;                // k / k2
    float* t1     = ws + 2 * ME;                // q / q2
    float* t2     = ws + 3 * ME;                // v / v2
    float* t3     = ws + 4 * ME;                // attn out / ffn out
    float* xb     = ws + 5 * ME;                // post-ln1
    float* xc     = ws + 6 * ME;                // post-ln2 (dec)
    float* db0    = ws + 7 * ME;                // dec x ping
    float* db1    = ws + 8 * ME;                // dec x pong
    float* kvbuf  = ws + 9 * ME;                // 5 layers of enc outputs
    float* sc     = ws + 14 * ME;               // (B, S, S) scores
    float* hid    = ws + 15 * ME;               // (MR, H) ffn hidden (4*ME)
    float* hN     = ws + 19 * ME;               // (B, E)

    const float scale = 1.0f / sqrtf((float)CE);
    const long long SE = (long long)CS * CE;
    const long long SS = (long long)CS * CS;

    // batched attention: out = softmax(scale * K Q^T) V
    auto attn = [&](const float* k, const float* q, const float* v, float* out) {
        launch_gemm(stream, k, q, nullptr, sc, CS, CS, CE, CB, SE, SE, SS, scale, false, false);
        softmax_rows_kernel<<<dim3(CB * CS), dim3(256), 0, stream>>>(sc, CS);
        launch_gemm(stream, sc, v, nullptr, out, CS, CE, CS, CB, SS, SE, SE, 1.0f, false, true);
    };

    // ---- embedding: (MR x V) @ (V x E) + b ----
    launch_gemm(stream, seq, embed_w, embed_b, embeds, MR, CE, CV, 1, 0, 0, 0, 1.0f, false, true);

    // ---- encoder ----
    const float* x = embeds;
    for (int l = 0; l < CL; ++l) {
        const long long oEE = (long long)l * CE * CE;
        const long long oEH = (long long)l * CE * CH;
        const long long oE  = (long long)l * CE;
        const long long oH  = (long long)l * CH;

        launch_gemm(stream, x, e_wk + oEE, e_bk + oE, t0, MR, CE, CE, 1, 0, 0, 0, 1.0f, false, true);
        launch_gemm(stream, x, e_wq + oEE, e_bq + oE, t1, MR, CE, CE, 1, 0, 0, 0, 1.0f, false, true);
        launch_gemm(stream, x, e_wv + oEE, e_bv + oE, t2, MR, CE, CE, 1, 0, 0, 0, 1.0f, false, true);
        attn(t0, t1, t2, t3);
        ln_residual_kernel<<<dim3(MR), dim3(256), 0, stream>>>(x, t3, e_ln1g + oE, e_ln1b + oE, xb, CE);

        launch_gemm(stream, xb, e_w1 + oEH, e_b1 + oH, hid, MR, CH, CE, 1, 0, 0, 0, 1.0f, true, true);
        launch_gemm(stream, hid, e_w2 + oEH, e_b2 + oE, t3, MR, CE, CH, 1, 0, 0, 0, 1.0f, true, true);

        float* out = kvbuf + (long long)l * ME;   // kept for decoder cross-attn
        ln_residual_kernel<<<dim3(MR), dim3(256), 0, stream>>>(xb, t3, e_ln2g + oE, e_ln2b + oE, out, CE);
        x = out;
    }

    // ---- decoder ----
    x = embeds;
    for (int l = 0; l < CL; ++l) {
        const long long oEE = (long long)l * CE * CE;
        const long long oEH = (long long)l * CE * CH;
        const long long oE  = (long long)l * CE;
        const long long oH  = (long long)l * CH;
        const float* kvl = kvbuf + (long long)l * ME;

        // self-attention
        launch_gemm(stream, x, d_wk + oEE, d_bk + oE, t0, MR, CE, CE, 1, 0, 0, 0, 1.0f, false, true);
        launch_gemm(stream, x, d_wq + oEE, d_bq + oE, t1, MR, CE, CE, 1, 0, 0, 0, 1.0f, false, true);
        launch_gemm(stream, x, d_wv + oEE, d_bv + oE, t2, MR, CE, CE, 1, 0, 0, 0, 1.0f, false, true);
        attn(t0, t1, t2, t3);
        ln_residual_kernel<<<dim3(MR), dim3(256), 0, stream>>>(x, t3, d_ln1g + oE, d_ln1b + oE, xb, CE);

        // cross-attention (k2,v2 from encoder layer output; q2 from xb)
        launch_gemm(stream, kvl, d_wk2 + oEE, d_bk2 + oE, t0, MR, CE, CE, 1, 0, 0, 0, 1.0f, false, true);
        launch_gemm(stream, xb,  d_wq2 + oEE, d_bq2 + oE, t1, MR, CE, CE, 1, 0, 0, 0, 1.0f, false, true);
        launch_gemm(stream, kvl, d_wv2 + oEE, d_bv2 + oE, t2, MR, CE, CE, 1, 0, 0, 0, 1.0f, false, true);
        attn(t0, t1, t2, t3);
        ln_residual_kernel<<<dim3(MR), dim3(256), 0, stream>>>(xb, t3, d_ln2g + oE, d_ln2b + oE, xc, CE);

        // FFN
        launch_gemm(stream, xc, d_w1 + oEH, d_b1 + oH, hid, MR, CH, CE, 1, 0, 0, 0, 1.0f, true, true);
        launch_gemm(stream, hid, d_w2 + oEH, d_b2 + oE, t3, MR, CE, CH, 1, 0, 0, 0, 1.0f, true, true);

        float* out = (l & 1) ? db1 : db0;
        ln_residual_kernel<<<dim3(MR), dim3(256), 0, stream>>>(xc, t3, d_ln3g + oE, d_ln3b + oE, out, CE);
        x = out;
    }

    // ---- head: softmax(hN @ unembed + b) ----
    gather_last_kernel<<<dim3((CB * CE + 255) / 256), dim3(256), 0, stream>>>(x, hN, CS, CE);
    launch_gemm(stream, hN, unembed_w, unembed_b, (float*)d_out, CB, CV, CE, 1, 0, 0, 0, 1.0f, false, true);
    softmax_rows_kernel<<<dim3(CB), dim3(256), 0, stream>>>((float*)d_out, CV);
}
